// perm_eq_linear_48756468744679
// MI455X (gfx1250) — matrix operand order repaired and checked by reading, not compiled, not run on hardware
//
#include <hip/hip_runtime.h>
#include <hip/hip_bf16.h>

typedef __attribute__((ext_vector_type(16))) _Float16 v16h;
typedef __attribute__((ext_vector_type(8)))  float    v8f;

#define N_NODES 1024
#define KNB 16
#define CI 32
#define CO 32

union AFrag { v16h v; _Float16 h[16]; };
union CFrag { v8f  v; float    f[8];  };

// weight indexing: arr[p][d][s] with d=CI, s=CO
#define WIDX(arr,p,d,s) arr[(((p)*CI + (d))*CO) + (s)]

// One workgroup (256 threads = 8 wave32) per node.
__global__ __launch_bounds__(256) void pe_node_kernel(
    const float* __restrict__ x1,  const float* __restrict__ A,
    const float* __restrict__ w22, const float* __restrict__ b22,
    const float* __restrict__ w21, const float* __restrict__ b21,
    const float* __restrict__ w12, const float* __restrict__ b12,
    const float* __restrict__ w11, const float* __restrict__ b11,
    const int*   __restrict__ nbrs,
    float* __restrict__ edge_sum, float* __restrict__ node_sum,
    float* __restrict__ ecnt,     float* __restrict__ ncnt)
{
    __shared__ int   s_nbr[KNB];
    __shared__ float s_adj[CI][KNB][KNB];   // 32 KB gathered sub-adjacency
    __shared__ float s_x[CI][KNB];
    __shared__ float s_sr[CI][KNB];         // row sums / 16
    __shared__ float s_sc[CI][KNB];         // col sums / 16
    __shared__ float s_dg[CI][KNB];         // diagonal
    __shared__ float s_sd[CI], s_sa[CI], s_xm[CI];
    // eq22 rank-1 pieces (i-varying, j-varying, diagonal, const)
    __shared__ float s_Rr[CO][KNB], s_Rc[CO][KNB], s_Dg[CO][KNB];
    // eq12 pieces
    __shared__ float s_Rr12[CO][KNB], s_Rc12[CO][KNB], s_Dg12[CO][KNB];
    __shared__ float s_cst[CO], s_c12[CO];
    __shared__ float s_xo[KNB][2*CO];       // per-node x_o [16][64]

    const int tid = threadIdx.x;
    const int b   = blockIdx.x;

    if (tid < KNB) s_nbr[tid] = nbrs[b*KNB + tid];
    __syncthreads();

    // ---- gather: adj_sub[d][i][j] = A[nbr[i], nbr[j], d] (d innermost -> coalesced) ----
    for (int t = tid; t < KNB*KNB*CI; t += 256) {
        int d  = t & (CI-1);
        int pr = t >> 5;            // (i,j) pair 0..255
        int i  = pr >> 4, j = pr & 15;
        size_t off = ((size_t)s_nbr[i]*N_NODES + (size_t)s_nbr[j])*CI + d;
        s_adj[d][i][j] = A[off];
    }
    for (int t = tid; t < KNB*CI; t += 256) {
        int d = t & (CI-1);
        int i = t >> 5;
        s_x[d][i] = x1[(size_t)s_nbr[i]*CI + d];
    }
    __syncthreads();

    // ---- per-channel reductions ----
    for (int t = tid; t < CI*KNB; t += 256) {
        int d = t >> 4, i = t & 15;
        float rs = 0.f, cs = 0.f;
        #pragma unroll
        for (int j = 0; j < KNB; ++j) { rs += s_adj[d][i][j]; cs += s_adj[d][j][i]; }
        s_sr[d][i] = rs * (1.f/16.f);
        s_sc[d][i] = cs * (1.f/16.f);
        s_dg[d][i] = s_adj[d][i][i];
    }
    __syncthreads();
    if (tid < CI) {
        int d = tid;
        float sds = 0.f, sas = 0.f, xm = 0.f;
        #pragma unroll
        for (int i = 0; i < KNB; ++i) { sds += s_dg[d][i]; sas += s_sr[d][i]; xm += s_x[d][i]; }
        s_sd[d] = sds*(1.f/16.f); s_sa[d] = sas*(1.f/16.f); s_xm[d] = xm*(1.f/16.f);
    }
    __syncthreads();

    // ---- rank-1 basis contractions (tiny GEMMs, plain VALU) ----
    for (int t = tid; t < CO*KNB; t += 256) {
        int s = t >> 4, i = t & 15;
        float rr=0,rc=0,dg=0,rr12=0,rc12=0,dg12=0,xo21=0,xo11=0;
        for (int d = 0; d < CI; ++d) {
            float scv=s_sc[d][i], srv=s_sr[d][i], dgv=s_dg[d][i], xv=s_x[d][i];
            float sdv=s_sd[d],    sav=s_sa[d],    xmv=s_xm[d];
            rr  += WIDX(w22,7,d,s)*scv + WIDX(w22,8,d,s)*srv + WIDX(w22,12,d,s)*dgv;
            rc  += WIDX(w22,5,d,s)*scv + WIDX(w22,6,d,s)*srv + WIDX(w22,11,d,s)*dgv;
            dg  += WIDX(w22,0,d,s)*dgv + WIDX(w22,2,d,s)*srv + WIDX(w22,3,d,s)*scv
                 + WIDX(w22,1,d,s)*sdv + WIDX(w22,4,d,s)*sav;
            rr12 += WIDX(w12,3,d,s)*xv;
            rc12 += WIDX(w12,2,d,s)*xv;
            dg12 += WIDX(w12,0,d,s)*xv + WIDX(w12,1,d,s)*xmv;
            xo21 += WIDX(w21,0,d,s)*dgv + WIDX(w21,1,d,s)*srv + WIDX(w21,2,d,s)*scv
                  + WIDX(w21,3,d,s)*sdv + WIDX(w21,4,d,s)*sav;
            xo11 += WIDX(w11,0,d,s)*xv + WIDX(w11,1,d,s)*xmv;
        }
        s_Rr[s][i]=rr;   s_Rc[s][i]=rc;   s_Dg[s][i]=dg;
        s_Rr12[s][i]=rr12; s_Rc12[s][i]=rc12; s_Dg12[s][i]=dg12;
        s_xo[i][s]    = xo21 + b21[s];
        s_xo[i][s+CO] = xo11 + b11[s];
        if (i == 0) {
            float cst = 0.f, c12 = 0.f;
            for (int d = 0; d < CI; ++d) {
                cst += WIDX(w22,13,d,s)*s_sd[d] + WIDX(w22,14,d,s)*s_sa[d];
                c12 += WIDX(w12,4,d,s)*s_xm[d];
            }
            s_cst[s] = cst + b22[s];
            s_c12[s] = c12 + b12[s];
        }
    }
    __syncthreads();

    // ---- counts ----
    {
        int p = tid, i = p >> 4, j = p & 15;      // 256 pixels, one per thread
        size_t eid = (size_t)s_nbr[i]*N_NODES + (size_t)s_nbr[j];
        atomicAdd(&ecnt[eid], 1.0f);
    }
    if (tid < KNB) atomicAdd(&ncnt[s_nbr[tid]], 1.0f);

    // ---- eq12/eq11 output channels 32..63 -> scatter edges ----
    for (int t = tid; t < 256*CO; t += 256) {
        int p = t >> 5, sch = t & 31;
        int i = p >> 4, j = p & 15;
        float v = s_Rr12[sch][i] + s_Rc12[sch][j]
                + ((i==j) ? s_Dg12[sch][i] : 0.f) + s_c12[sch];
        size_t eid = (size_t)s_nbr[i]*N_NODES + (size_t)s_nbr[j];
        atomicAdd(&edge_sum[eid*64 + 32 + sch], v);
    }

    // ---- node scatter ----
    for (int t = tid; t < KNB*2*CO; t += 256) {
        int j = t >> 6, sch = t & 63;
        atomicAdd(&node_sum[(size_t)s_nbr[j]*64 + sch], s_xo[j][sch]);
    }

    // ---- WMMA: out[p,s] += sum_d adj[d,i,j]*w22[9,d,s] + adj[d,j,i]*w22[10,d,s] ----
    // Per-node GEMM [256 x 32] x [32 x 32] (x2 for adjT); 8 waves x 2 M-tiles each.
    const int lane = tid & 31;
    const int wv   = tid >> 5;      // wave 0..7
    const int hi   = lane >> 4;     // K-half select
    const int lr   = lane & 15;     // M (A) / N (B,C) row

    // B fragments (16-bit B 32x16 layout: VGPR v, elem t -> K = 2*(v&3)+t + (v>=4?16:0) + (lane>=16?8:0))
    AFrag bf[2][2];
    #pragma unroll
    for (int w = 0; w < 2; ++w)
        #pragma unroll
        for (int nt = 0; nt < 2; ++nt)
            #pragma unroll
            for (int e = 0; e < 16; ++e) {
                int vv = e >> 1, tt = e & 1;
                int k  = ((vv & 3) << 1) + tt + ((vv >> 2) << 4) + (hi << 3);          // d index 0..31
                bf[w][nt].h[e] = (_Float16)WIDX(w22, 9+w, k, lr + 16*nt);
            }

    #pragma unroll
    for (int mtl = 0; mtl < 2; ++mtl) {
        int mt = wv*2 + mtl;          // M-tile 0..15
        int p  = mt*16 + lr;          // pixel for this lane's A row
        int i  = p >> 4, j = p & 15;
        // 16-bit A 16x32 layout: VGPR v(0..7), elem t -> K = 2*(v&3)+t + (v>=4?16:0) + (lane>=16?8:0)
        AFrag an, at;
        #pragma unroll
        for (int e = 0; e < 16; ++e) {
            int vv = e >> 1, tt = e & 1;
            int k  = ((vv & 3) << 1) + tt + ((vv >> 2) << 4) + (hi << 3);
            an.h[e] = (_Float16)s_adj[k][i][j];
            at.h[e] = (_Float16)s_adj[k][j][i];
        }
        #pragma unroll
        for (int nt = 0; nt < 2; ++nt) {
            CFrag c;
            #pragma unroll
            for (int r = 0; r < 8; ++r) c.f[r] = 0.f;
            c.v = __builtin_amdgcn_wmma_f32_16x16x32_f16(
                      false, an.v, false, bf[0][nt].v, (short)0, c.v, false, false);
            c.v = __builtin_amdgcn_wmma_f32_16x16x32_f16(
                      false, at.v, false, bf[1][nt].v, (short)0, c.v, false, false);
            int sch = lr + 16*nt;
            // C layout: VGPR r, lane -> M = r + (lane>=16 ? 8 : 0), N = lane&15
            #pragma unroll
            for (int r = 0; r < 8; ++r) {
                int pm = mt*16 + r + (hi ? 8 : 0);
                int ii = pm >> 4, jj = pm & 15;
                float val = c.f[r] + s_Rr[sch][ii] + s_Rc[sch][jj]
                          + ((ii==jj) ? s_Dg[sch][ii] : 0.f) + s_cst[sch];
                size_t eid = (size_t)s_nbr[ii]*N_NODES + (size_t)s_nbr[jj];
                atomicAdd(&edge_sum[eid*64 + sch], val);
            }
        }
    }
}

// mean + ELU over edge rows then node rows (node section directly follows edges in d_out)
__global__ __launch_bounds__(256) void pe_finalize_kernel(
    float* __restrict__ out, const float* __restrict__ ecnt, const float* __restrict__ ncnt)
{
    size_t g = (size_t)blockIdx.x * blockDim.x + threadIdx.x;
    const size_t nEdges = (size_t)N_NODES * N_NODES;
    if (g >= nEdges + N_NODES) return;
    float c = (g < nEdges) ? ecnt[g] : ncnt[g - nEdges];
    if (c < 0.5f) return;                 // untouched rows stay 0; elu(0)==0
    float inv = 1.0f / c;                 // c >= 1 here => max(c,1)==c
    float* row = out + g * 64;
    #pragma unroll 4
    for (int s = 0; s < 64; ++s) {
        float v = row[s] * inv;
        row[s] = (v > 0.f) ? v : expm1f(v);
    }
}

extern "C" void kernel_launch(void* const* d_in, const int* in_sizes, int n_in,
                              void* d_out, int out_size, void* d_ws, size_t ws_size,
                              hipStream_t stream) {
    (void)in_sizes; (void)n_in; (void)ws_size;
    const float* x1  = (const float*)d_in[0];
    const float* A   = (const float*)d_in[1];
    const float* w22 = (const float*)d_in[2];
    const float* b22 = (const float*)d_in[3];
    const float* w21 = (const float*)d_in[4];
    const float* b21 = (const float*)d_in[5];
    const float* w12 = (const float*)d_in[6];
    const float* b12 = (const float*)d_in[7];
    const float* w11 = (const float*)d_in[8];
    const float* b11 = (const float*)d_in[9];
    const int*   nbrs= (const int*)d_in[10];

    float* out  = (float*)d_out;
    float* ecnt = (float*)d_ws;                               // [N*N] counts
    float* ncnt = ecnt + (size_t)N_NODES*N_NODES;             // [N] counts

    // zero accumulators every call (graph-capture-safe memset nodes)
    hipMemsetAsync(d_out, 0, (size_t)out_size * sizeof(float), stream);
    hipMemsetAsync(d_ws,  0, ((size_t)N_NODES*N_NODES + N_NODES) * sizeof(float), stream);

    float* edge_sum = out;                                    // [N*N, 64]
    float* node_sum = out + (size_t)N_NODES*N_NODES*64;       // [N, 64]

    pe_node_kernel<<<N_NODES, 256, 0, stream>>>(
        x1, A, w22, b22, w21, b21, w12, b12, w11, b11, nbrs,
        edge_sum, node_sum, ecnt, ncnt);

    size_t total = (size_t)N_NODES*N_NODES + N_NODES;
    int blocks = (int)((total + 255) / 256);
    pe_finalize_kernel<<<blocks, 256, 0, stream>>>(out, ecnt, ncnt);
}
